// IndexConv1D_75831942578229
// MI455X (gfx1250) — matrix operand order repaired and checked by reading, not compiled, not run on hardware
//
#include <hip/hip_runtime.h>
#include <hip/hip_bf16.h>

typedef __attribute__((ext_vector_type(16))) __bf16 v16bf;
typedef __attribute__((ext_vector_type(8)))  __bf16 v8bf;
typedef __attribute__((ext_vector_type(8)))  float  v8f;

#define FDIM 96      // output features per column
#define NCOLS 3
#define KTAPS 3
#define TM 32        // output rows per block
#define ROWS (TM + 2)
#define BLOCK 384    // 12 waves: 2 M-subtiles x 6 N-subtiles

// ---- bf16 hi/lo split: x ~= (float)hi + (float)lo (~16 mantissa bits) ----
__device__ __forceinline__ void bf16_split(float x, __bf16& hi, __bf16& lo) {
    hi = (__bf16)x;
    lo = (__bf16)(x - (float)hi);
}

// Pre-swizzle fp32 weights (3*CIN, 96) into fragment-ready bf16 hi/lo arrays.
// Linear layout: idx = ((ch*6 + t)*32 + lane)*16 + e
//   value = W[(ch*32 + 8*(lane>=16) + e + 8*(e>=8)) * 96 + t*16 + (lane&15)]
// so a B fragment for (chunk ch, tile t) is 32 contiguous bytes per lane.
__global__ __launch_bounds__(256)
void swizzle_weights(const float* __restrict__ W,
                     __bf16* __restrict__ Whi, __bf16* __restrict__ Wlo,
                     int kc_total) {
    const int idx = blockIdx.x * 256 + threadIdx.x;
    if (idx >= kc_total * FDIM) return;
    const int e    = idx & 15;
    const int lane = (idx >> 4) & 31;
    const int ct   = idx >> 9;            // ch*6 + t
    const int t    = ct % 6;
    const int ch   = ct / 6;
    const int krow = ch * 32 + ((lane < 16) ? 0 : 8) + e + ((e < 8) ? 0 : 8);
    const int col  = t * 16 + (lane & 15);
    __bf16 h, l;
    bf16_split(W[(size_t)krow * FDIM + col], h, l);
    Whi[idx] = h;
    Wlo[idx] = l;
}

// A fragment (16x32 bf16) from the bf16 LDS tile: two aligned b128 ds loads.
// Wave32 A layout: lanes 0-15 = rows M holding K {0..7,16..23};
// lanes 16-31 = same rows holding K {8..15,24..31}.
__device__ __forceinline__ v16bf load_a_frag(const __bf16* __restrict__ s, int ldr,
                                             int r0, int tap, int c0, int lane) {
    const int row   = r0 + (lane & 15) + tap;       // LDS row (r=0 <=> n0-1)
    const int kbase = (lane < 16) ? 0 : 8;
    const __bf16* base = s + row * ldr + c0 + kbase; // 16B aligned
    const v8bf p0 = *(const v8bf*)(base);
    const v8bf p1 = *(const v8bf*)(base + 16);
    return __builtin_shufflevector(p0, p1, 0, 1, 2, 3, 4, 5, 6, 7,
                                   8, 9, 10, 11, 12, 13, 14, 15);
}

// Gather -> K=3 SAME conv (GEMM vs (3*CIN,96)) -> relu(+bias) -> scatter.
// resid != nullptr adds the residual of the second _infer.
template <int CIN>
__global__ __launch_bounds__(BLOCK)
void conv_gather_wmma(const float*  __restrict__ in,     // (N, CIN)
                      const int*    __restrict__ index,  // (N, 3)
                      const __bf16* __restrict__ Whi,    // swizzled hi
                      const __bf16* __restrict__ Wlo,    // swizzled lo
                      const float*  __restrict__ bias,   // (96)
                      const float*  __restrict__ resid,  // (N, 288) or nullptr
                      float* __restrict__ out,           // (N, 288)
                      int n_points) {
    __shared__ __bf16 s_hi[ROWS * CIN];
    __shared__ __bf16 s_lo[ROWS * CIN];
    __shared__ int    s_idx[ROWS];

    const int ntiles = (n_points + TM - 1) / TM;
    const int j      = blockIdx.x / ntiles;     // index column 0..2
    const int n0     = (blockIdx.x % ntiles) * TM;
    const int tid    = threadIdx.x;

    // Stage gather indices for rows n0-1 .. n0+TM (SAME padding => -1 sentinel).
    for (int r = tid; r < ROWS; r += BLOCK) {
        const int m = n0 - 1 + r;
        s_idx[r] = (m >= 0 && m < n_points) ? index[m * NCOLS + j] : -1;
    }
    __syncthreads();
    // Stage gathered rows, splitting fp32 -> bf16 hi/lo exactly once per element.
    for (int i = tid; i < ROWS * CIN; i += BLOCK) {
        const int r = i / CIN, c = i - r * CIN;
        const int g = s_idx[r];
        const float x = (g >= 0) ? in[(size_t)g * CIN + c] : 0.0f;
        __bf16 h, l;
        bf16_split(x, h, l);
        s_hi[i] = h;
        s_lo[i] = l;
    }
    __syncthreads();

    const int wave = tid >> 5;         // 0..11
    const int lane = tid & 31;
    const int r0   = (wave & 1) * 16;  // M-subtile
    const int t    = wave >> 1;        // N-subtile 0..5 (cols t*16..t*16+15)

    v8f acc = {};
    constexpr int KCHUNKS = (KTAPS * CIN) / 32;
    // Per-wave pointer into the swizzled weights: 32B per lane per chunk/tile.
    const __bf16* wh = Whi + (((size_t)t * 32 + lane) << 4);
    const __bf16* wl = Wlo + (((size_t)t * 32 + lane) << 4);
#pragma unroll 3
    for (int ch = 0; ch < KCHUNKS; ++ch) {
        const int kc0 = ch * 32;
        const int tap = kc0 / CIN;
        const int c0  = kc0 - tap * CIN;
        const v16bf ahi = load_a_frag(s_hi, CIN, r0, tap, c0, lane);
        const v16bf alo = load_a_frag(s_lo, CIN, r0, tap, c0, lane);
        const v16bf bhi = *(const v16bf*)(wh + (size_t)ch * (6 * 512));
        const v16bf blo = *(const v16bf*)(wl + (size_t)ch * (6 * 512));
        // Compensated bf16x2: hi*hi + hi*lo + lo*hi (error ~ 2^-16 rel)
        acc = __builtin_amdgcn_wmma_f32_16x16x32_bf16(false, ahi, false, bhi,
                                                      (short)0, acc, false, false);
        acc = __builtin_amdgcn_wmma_f32_16x16x32_bf16(false, ahi, false, blo,
                                                      (short)0, acc, false, false);
        acc = __builtin_amdgcn_wmma_f32_16x16x32_bf16(false, alo, false, bhi,
                                                      (short)0, acc, false, false);
    }

    // C/D layout: VGPR d -> M = d + 8*(lane>=16), N = lane&15.
    const int f    = t * 16 + (lane & 15);
    const float bv = bias[f];
    const int mofs = (lane < 16) ? 0 : 8;
#pragma unroll
    for (int d = 0; d < 8; ++d) {
        const int m = r0 + d + mofs;
        const int n = n0 + m;
        if (n < n_points) {
            const int g = s_idx[m + 1];                 // index[n, j]
            float y = acc[d] + bv;
            y = y > 0.0f ? y : 0.0f;
            const size_t off = (size_t)g * (NCOLS * FDIM) + j * FDIM + f;
            out[off] = resid ? resid[off] + y : y;      // scatter (permutation)
        }
    }
}

// In-place LayerNorm over 288 features; one wave32 per row, 9 vals/lane.
__global__ __launch_bounds__(256)
void layernorm288(float* __restrict__ h, const float* __restrict__ gamma,
                  const float* __restrict__ beta, int n_points) {
    const int lane = threadIdx.x & 31;
    const int row  = blockIdx.x * 8 + (threadIdx.x >> 5);
    if (row >= n_points) return;
    float* p = h + (size_t)row * 288;
    float v[9], s = 0.0f, s2 = 0.0f;
#pragma unroll
    for (int i = 0; i < 9; ++i) {
        v[i] = p[lane + i * 32];
        s += v[i]; s2 += v[i] * v[i];
    }
#pragma unroll
    for (int m = 16; m >= 1; m >>= 1) {
        s  += __shfl_xor(s,  m, 32);
        s2 += __shfl_xor(s2, m, 32);
    }
    const float mu   = s * (1.0f / 288.0f);
    const float var  = s2 * (1.0f / 288.0f) - mu * mu;
    const float rstd = __frsqrt_rn(var + 1.0e-3f);
#pragma unroll
    for (int i = 0; i < 9; ++i) {
        const int c = lane + i * 32;
        p[c] = (v[i] - mu) * rstd * gamma[c] + beta[c];
    }
}

extern "C" void kernel_launch(void* const* d_in, const int* in_sizes, int n_in,
                              void* d_out, int out_size, void* d_ws, size_t ws_size,
                              hipStream_t stream) {
    const float* X     = (const float*)d_in[0];
    const int*   index = (const int*)  d_in[1];
    const float* W1    = (const float*)d_in[2];  // (3,96,96)  -> (288,96)
    const float* b1    = (const float*)d_in[3];
    const float* W2    = (const float*)d_in[4];  // (3,288,96) -> (864,96)
    const float* b2    = (const float*)d_in[5];
    const float* gamma = (const float*)d_in[6];
    const float* beta  = (const float*)d_in[7];

    const int n = in_sizes[0] / 96;              // N (X is (1,N,96))
    float* out = (float*)d_out;                  // first n*288 floats

    // Workspace carve-up (256B aligned): h1, then swizzled weights.
    auto align256 = [](size_t x) { return (x + 255) & ~(size_t)255; };
    char* ws = (char*)d_ws;
    size_t off = 0;
    float* h1 = (float*)(ws + off);  off = align256(off + (size_t)n * 288 * sizeof(float));
    const int kc1 = KTAPS * 96, kc2 = KTAPS * 288;
    __bf16* W1hi = (__bf16*)(ws + off); off = align256(off + (size_t)kc1 * FDIM * 2);
    __bf16* W1lo = (__bf16*)(ws + off); off = align256(off + (size_t)kc1 * FDIM * 2);
    __bf16* W2hi = (__bf16*)(ws + off); off = align256(off + (size_t)kc2 * FDIM * 2);
    __bf16* W2lo = (__bf16*)(ws + off); off = align256(off + (size_t)kc2 * FDIM * 2);

    // Pre-swizzle weights into WMMA B-fragment layout (bf16 hi/lo).
    swizzle_weights<<<dim3((kc1 * FDIM + 255) / 256), dim3(256), 0, stream>>>(W1, W1hi, W1lo, kc1);
    swizzle_weights<<<dim3((kc2 * FDIM + 255) / 256), dim3(256), 0, stream>>>(W2, W2hi, W2lo, kc2);

    const int ntiles = (n + TM - 1) / TM;
    dim3 grid(ntiles * NCOLS), block(BLOCK);

    // h1 = _infer(X, index, W1, b1)            (every element written: perm scatter)
    conv_gather_wmma<96><<<grid, block, 0, stream>>>(X, index, W1hi, W1lo, b1, nullptr, h1, n);
    // out = h1 + _infer(h1, index, W2, b2)
    conv_gather_wmma<288><<<grid, block, 0, stream>>>(h1, index, W2hi, W2lo, b2, h1, out, n);
    // out = layernorm(out) * gamma + beta      (in place)
    layernorm288<<<dim3((n + 7) / 8), dim3(256), 0, stream>>>(out, gamma, beta, n);

    // Second tuple element: the int32 index, appended raw after the float output.
    hipMemcpyAsync((char*)d_out + (size_t)n * 288 * sizeof(float), d_in[1],
                   (size_t)n * NCOLS * sizeof(int), hipMemcpyDeviceToDevice, stream);
}
